// Attention_42322607735501
// MI455X (gfx1250) — matrix operand order repaired and checked by reading, not compiled, not run on hardware
//
#include <hip/hip_runtime.h>
#include <math.h>

typedef __attribute__((ext_vector_type(16))) __bf16 bf16x16;
typedef __attribute__((ext_vector_type(8)))  float  v8f;

#define B_ 32
#define S_ 2048
#define E_ 1024
#define Q_ 1024
#define A_ 512
#define M_ (B_ * S_)   // 65536

// GEMM tiling
#define BM 128
#define BN 128
#define BK 32
#define LDA 40   // padded LDS row stride in bf16 (80 bytes, 16B-aligned)

struct __align__(16) F4 { float x, y, z, w; };
struct __align__(16) Q4 { unsigned int x, y, z, w; };
union FragU { Q4 q[2]; bf16x16 v; __bf16 h[16]; };

// ---------------- proj_q = query @ W_q : [B,Q] x [Q,A] -> [B,A] ----------------
__global__ __launch_bounds__(256) void projq_kernel(const float* __restrict__ query,
                                                    const float* __restrict__ Wq,
                                                    float* __restrict__ projq) {
    const int a = blockIdx.x * 256 + threadIdx.x;   // 0..A_-1
    const int b = blockIdx.y;
    const float* q = query + b * Q_;
    float acc = 0.f;
    for (int k = 0; k < Q_; ++k)
        acc += q[k] * Wq[k * A_ + a];
    projq[b * A_ + a] = acc;
}

// ---------------- Wt[a][e] = bf16(W_in[e][a]) (transpose + convert) ----------------
__global__ __launch_bounds__(256) void convert_wt(const float* __restrict__ Win,
                                                  __bf16* __restrict__ Wt) {
    const int idx = blockIdx.x * 256 + threadIdx.x;  // 0 .. E*A-1, a fastest (coalesced read)
    const int e = idx / A_;
    const int a = idx - e * A_;
    Wt[a * E_ + e] = (__bf16)Win[idx];
}

// ---------------- proj_in = inputs @ W_in, via bf16 WMMA, f32 accumulate ----------------
__global__ __launch_bounds__(256) void gemm_kernel(const float* __restrict__ X,    // [M_, E_] f32
                                                   const __bf16* __restrict__ Wt,  // [A_, E_] bf16
                                                   float* __restrict__ P) {        // [M_, A_] f32
    __shared__ __bf16 lds_a[BM * LDA];
    __shared__ __bf16 lds_b[BN * LDA];

    const int tid  = threadIdx.x;
    const int lane = tid & 31;
    const int wave = tid >> 5;      // 0..7
    const int wm   = wave >> 2;     // 0..1  (M direction)
    const int wn   = wave & 3;      // 0..3  (N direction)
    const int m0   = blockIdx.y * BM;
    const int n0   = blockIdx.x * BN;

    // Staging assignment: each thread owns row (tid>>1) of the 128-row tile,
    // and half (tid&1) of the 32-wide K slice (16 elements = 32B of bf16).
    const int sr = tid >> 1;               // 0..127
    const int sk = (tid & 1) * 16;         // 0 or 16
    const float* aptr = X + (size_t)(m0 + sr) * E_ + sk;

    // Async B staging: GVS addressing (SADDR = Wt, VADDR = 32-bit byte offset).
    // LDS destination VGPR holds the wave-relative LDS byte offset.
    const unsigned ldsb_addr = (unsigned)(unsigned long long)&lds_b[sr * LDA + sk];
    const unsigned boff_base = (unsigned)(((n0 + sr) * E_ + sk) * 2);

    const v8f zero = {0.f, 0.f, 0.f, 0.f, 0.f, 0.f, 0.f, 0.f};
    v8f acc[4][2];
#pragma unroll
    for (int mi = 0; mi < 4; ++mi)
#pragma unroll
        for (int ni = 0; ni < 2; ++ni) acc[mi][ni] = zero;

    const int lh = lane >> 4;   // lane half (0/1)
    const int ln = lane & 15;

    for (int k0 = 0; k0 < E_; k0 += BK) {
        // ---- stage B: async global->LDS (raw bf16 bytes, 2 x 16B per thread) ----
        {
            const unsigned go = boff_base + (unsigned)(k0 * 2);
            asm volatile(
                "global_load_async_to_lds_b128 %0, %1, %2 offset:0\n\t"
                "global_load_async_to_lds_b128 %0, %1, %2 offset:16"
                :: "v"(ldsb_addr), "v"(go), "s"(Wt)
                : "memory");
        }
        // ---- prefetch next A slice while we convert/compute ----
        if (k0 + BK < E_)
            __builtin_prefetch((const void*)(aptr + k0 + BK), 0, 3);
        // ---- stage A: 16 f32 -> 16 bf16 into LDS ----
        {
            const F4* s4 = (const F4*)(aptr + k0);
            F4 f0 = s4[0], f1 = s4[1], f2 = s4[2], f3 = s4[3];
            FragU pk;
            pk.h[0]  = (__bf16)f0.x; pk.h[1]  = (__bf16)f0.y; pk.h[2]  = (__bf16)f0.z; pk.h[3]  = (__bf16)f0.w;
            pk.h[4]  = (__bf16)f1.x; pk.h[5]  = (__bf16)f1.y; pk.h[6]  = (__bf16)f1.z; pk.h[7]  = (__bf16)f1.w;
            pk.h[8]  = (__bf16)f2.x; pk.h[9]  = (__bf16)f2.y; pk.h[10] = (__bf16)f2.z; pk.h[11] = (__bf16)f2.w;
            pk.h[12] = (__bf16)f3.x; pk.h[13] = (__bf16)f3.y; pk.h[14] = (__bf16)f3.z; pk.h[15] = (__bf16)f3.w;
            Q4* dst = (Q4*)&lds_a[sr * LDA + sk];
            dst[0] = pk.q[0]; dst[1] = pk.q[1];
        }
        // Our inline-asm async ops are invisible to the compiler's counters:
        // drain ASYNCcnt explicitly before the workgroup barrier.
        asm volatile("s_wait_asynccnt 0x0" ::: "memory");
        __syncthreads();

        // ---- load fragments per ISA VGPR layouts ----
        bf16x16 afrag[4];
#pragma unroll
        for (int mi = 0; mi < 4; ++mi) {
            // A 16x32 bf16: lane holds row M=ln; K = lh*8 + 0..7 and 16 + lh*8 + 0..7
            const int row = wm * 64 + mi * 16 + ln;
            FragU u;
            u.q[0] = *(const Q4*)&lds_a[row * LDA + lh * 8];
            u.q[1] = *(const Q4*)&lds_a[row * LDA + 16 + lh * 8];
            afrag[mi] = u.v;
        }
        bf16x16 bfrag[2];
#pragma unroll
        for (int ni = 0; ni < 2; ++ni) {
            // B 32x16 bf16 (column-major in LDS): lane holds col N=ln; K = lh*8 + 0..7 and 16 + lh*8 + 0..7
            const int n = wn * 32 + ni * 16 + ln;
            FragU u;
            u.q[0] = *(const Q4*)&lds_b[n * LDA + lh * 8];
            u.q[1] = *(const Q4*)&lds_b[n * LDA + 16 + lh * 8];
            bfrag[ni] = u.v;
        }

#pragma unroll
        for (int mi = 0; mi < 4; ++mi)
#pragma unroll
            for (int ni = 0; ni < 2; ++ni)
                acc[mi][ni] = __builtin_amdgcn_wmma_f32_16x16x32_bf16(
                    false, afrag[mi], false, bfrag[ni],
                    (short)0, acc[mi][ni], false, false);

        __syncthreads();
    }

    // ---- epilogue: C layout -> row = v + 8*lh, col = ln ----
#pragma unroll
    for (int mi = 0; mi < 4; ++mi)
#pragma unroll
        for (int ni = 0; ni < 2; ++ni) {
            const int col   = n0 + wn * 32 + ni * 16 + ln;
            const int rbase = m0 + wm * 64 + mi * 16 + lh * 8;
#pragma unroll
            for (int v = 0; v < 8; ++v)
                P[(size_t)(rbase + v) * A_ + col] = acc[mi][ni][v];
        }
}

// ---------------- scores[b,s] = sum_a tanh(proj + proj_q) * w_att ----------------
__global__ __launch_bounds__(256) void score_kernel(const float* __restrict__ P,
                                                    const float* __restrict__ projq,
                                                    const float* __restrict__ watt,
                                                    float* __restrict__ scores) {
    const int row  = blockIdx.x * 8 + (threadIdx.x >> 5);  // one wave32 per (b,s) row
    const int lane = threadIdx.x & 31;
    const int b    = row >> 11;   // row / S_
    const float* prow = P + (size_t)row * A_;
    const float* qrow = projq + b * A_;
    float sum = 0.f;
    for (int c = lane; c < A_; c += 32)
        sum += tanhf(prow[c] + qrow[c]) * watt[c];
#pragma unroll
    for (int off = 16; off > 0; off >>= 1)
        sum += __shfl_xor(sum, off, 32);
    if (lane == 0) scores[row] = sum;
}

// ---------------- softmax over S, in place (scores -> weights) ----------------
__global__ __launch_bounds__(256) void softmax_kernel(float* __restrict__ scores) {
    __shared__ float red[256];
    const int b = blockIdx.x, t = threadIdx.x;
    float* sc = scores + b * S_;
    float v[8];
    float mx = -INFINITY;
#pragma unroll
    for (int j = 0; j < 8; ++j) { v[j] = sc[t + 256 * j]; mx = fmaxf(mx, v[j]); }
    red[t] = mx; __syncthreads();
    for (int s2 = 128; s2 > 0; s2 >>= 1) {
        if (t < s2) red[t] = fmaxf(red[t], red[t + s2]);
        __syncthreads();
    }
    mx = red[0]; __syncthreads();
    float sm = 0.f;
#pragma unroll
    for (int j = 0; j < 8; ++j) { v[j] = expf(v[j] - mx); sm += v[j]; }
    red[t] = sm; __syncthreads();
    for (int s2 = 128; s2 > 0; s2 >>= 1) {
        if (t < s2) red[t] += red[t + s2];
        __syncthreads();
    }
    const float inv = 1.f / red[0];
#pragma unroll
    for (int j = 0; j < 8; ++j) sc[t + 256 * j] = v[j] * inv;
}

// ---------------- context[b,a] = sum_s weights[b,s] * proj_in[b,s,a] ----------------
__global__ __launch_bounds__(256) void context_kernel(const float* __restrict__ P,
                                                      const float* __restrict__ weights,
                                                      float* __restrict__ out) {
    const int b = blockIdx.y;
    const int a = blockIdx.x * 256 + threadIdx.x;
    const float* w = weights + b * S_;
    const float* p = P + (size_t)b * S_ * A_ + a;
    float acc = 0.f;
    for (int s = 0; s < S_; ++s)
        acc += w[s] * p[(size_t)s * A_];
    out[b * A_ + a] = acc;
}

extern "C" void kernel_launch(void* const* d_in, const int* in_sizes, int n_in,
                              void* d_out, int out_size, void* d_ws, size_t ws_size,
                              hipStream_t stream) {
    const float* inputs = (const float*)d_in[0];
    const float* query  = (const float*)d_in[1];
    const float* W_in   = (const float*)d_in[2];
    const float* W_q    = (const float*)d_in[3];
    const float* w_att  = (const float*)d_in[4];
    float* out = (float*)d_out;

    float* ws     = (float*)d_ws;
    float* proj   = ws;                          // M_*A_ floats (128 MB)
    float* projq  = proj + (size_t)M_ * A_;      // B_*A_
    float* scores = projq + B_ * A_;             // B_*S_  (becomes weights in place)
    __bf16* Wt    = (__bf16*)(scores + B_ * S_); // A_*E_ bf16 (1 MB), 16B-aligned offset

    projq_kernel  <<<dim3(A_ / 256, B_), 256, 0, stream>>>(query, W_q, projq);
    convert_wt    <<<(E_ * A_) / 256,     256, 0, stream>>>(W_in, Wt);
    gemm_kernel   <<<dim3(A_ / BN, M_ / BM), 256, 0, stream>>>(inputs, Wt, proj);
    score_kernel  <<<M_ / 8,              256, 0, stream>>>(proj, projq, w_att, scores);
    softmax_kernel<<<B_,                  256, 0, stream>>>(scores);
    context_kernel<<<dim3(A_ / 256, B_),  256, 0, stream>>>(proj, scores, out);

    (void)in_sizes; (void)n_in; (void)out_size; (void)ws_size;
}
